// CausalSelfAttention_9199819948366
// MI455X (gfx1250) — compile-verified, matrix operand order checked by reading, not run on hardware
//
#include <hip/hip_runtime.h>

// ---------------------------------------------------------------------------
// CDNA5 (gfx1250) causal self-attention: bf16 WMMA pipeline, 128-bit fragment
// feeds everywhere.  B=4, T=2048, C=1024, H=16, D=64
// ---------------------------------------------------------------------------

typedef __attribute__((ext_vector_type(16))) __bf16 v16bf;
typedef __attribute__((ext_vector_type(8)))  float  v8f;

#define NB   4
#define NT   2048
#define NC   1024
#define NH   16
#define ND   64
#define NM   (NB * NT)      // 8192 rows

static __device__ __forceinline__ unsigned short f2bf_bits(float f) {
    unsigned int u = __builtin_bit_cast(unsigned int, f);
    unsigned int r = u + 0x7FFFu + ((u >> 16) & 1u);   // round-to-nearest-even
    return (unsigned short)(r >> 16);
}
static __device__ __forceinline__ __bf16 f2bf(float f) {
    return __builtin_bit_cast(__bf16, f2bf_bits(f));
}
static __device__ __forceinline__ unsigned int f2bf_pack(float lo, float hi) {
    return (unsigned int)f2bf_bits(lo) | ((unsigned int)f2bf_bits(hi) << 16);
}

// 16-bit A/B fragment (ISA 7.12.2): per lane the 16 elements sit at k offsets
// {8*hi .. 8*hi+7} and {16+8*hi .. 23+8*hi} -> two aligned 16-byte runs.
union FragU { uint4 q[2]; unsigned int u[8]; v16bf v; };

static __device__ __forceinline__ v16bf frag_k16(const __bf16* base, int hi) {
    FragU f;
    f.q[0] = *(const uint4*)(base + 8 * hi);
    f.q[1] = *(const uint4*)(base + 16 + 8 * hi);
    return f.v;
}

// ---------------------------------------------------------------------------
// f32 -> bf16 conversion (element-wise)
// ---------------------------------------------------------------------------
__global__ __launch_bounds__(256) void cvt_bf16_kernel(const float* __restrict__ in,
                                                       __bf16* __restrict__ out, int n) {
    int i = blockIdx.x * blockDim.x + threadIdx.x;
    if (i < n) out[i] = f2bf(in[i]);
}

// ---------------------------------------------------------------------------
// f32 [K x N] -> bf16 [N x K] transpose-convert (LDS tiled, 32x32)
// ---------------------------------------------------------------------------
__global__ __launch_bounds__(256) void cvt_transpose_kernel(const float* __restrict__ in,
                                                            __bf16* __restrict__ out,
                                                            int K, int N) {
    __shared__ float t[32][33];
    const int kb = blockIdx.x * 32, nb = blockIdx.y * 32;
    const int tx = threadIdx.x & 31, ty = threadIdx.x >> 5;   // 32 x 8
#pragma unroll
    for (int j = 0; j < 4; ++j) {
        int r = ty + j * 8;
        t[r][tx] = in[(size_t)(kb + r) * N + nb + tx];
    }
    __syncthreads();
#pragma unroll
    for (int j = 0; j < 4; ++j) {
        int r = ty + j * 8;
        out[(size_t)(nb + r) * K + kb + tx] = f2bf(t[tx][r]);
    }
}

// ---------------------------------------------------------------------------
// bf16 GEMM: C[8192 x N] = A[8192 x 1024] * W[1024 x N], W given TRANSPOSED
// ([N x 1024] bf16, streamed from L2).  Block tile 128x64, 8 waves of 32x32,
// K-step 32.  A tile LDS-staged (ds_load_b128 fragments), B fragments are two
// global_load_b128 each.
//   mode 0: scatter into Q[B,H,T,D] / K[B,H,T,D] / V^T[B,H,D,T] bf16
//   mode 1: fp32 out + bias
// ---------------------------------------------------------------------------
#define BM 128
#define BN 64
#define BK 32
#define LDA 40

__global__ __launch_bounds__(256) void gemm_bf16_kernel(
        const __bf16* __restrict__ A, const __bf16* __restrict__ Wt, int mode,
        __bf16* __restrict__ qb, __bf16* __restrict__ kb, __bf16* __restrict__ vt,
        float* __restrict__ outf, const float* __restrict__ bias) {
    __shared__ unsigned short sAu[BM * LDA];
    __bf16* sA = (__bf16*)sAu;

    const int tid  = threadIdx.x;
    const int wid  = tid >> 5;
    const int lid  = tid & 31;
    const int widM = wid >> 1;         // 0..3
    const int widN = wid & 1;          // 0..1
    const int nl   = lid & 15;
    const int hi   = lid >> 4;

    const int m0 = blockIdx.y * BM;
    const int n0 = blockIdx.x * BN;

    v8f acc[2][2] = {};

    for (int k0 = 0; k0 < NC; k0 += BK) {
        __syncthreads();
        {   // A tile: 128 rows x 32 cols, 2 x uint4 per thread
            int row = tid >> 1, half = tid & 1;
            const uint4* src = (const uint4*)(A + (size_t)(m0 + row) * NC + k0 + half * 16);
            uint4* dst = (uint4*)(sA + row * LDA + half * 16);
            dst[0] = src[0];
            dst[1] = src[1];
        }
        __syncthreads();

        if (k0 + BK < NC)   // prefetch next A tile -> global_prefetch_b8
            __builtin_prefetch(A + (size_t)(m0 + (tid >> 1)) * NC + k0 + BK, 0, 0);

        v16bf af[2], bfg[2];
#pragma unroll
        for (int rb = 0; rb < 2; ++rb)
            af[rb] = frag_k16(sA + (widM * 32 + rb * 16 + nl) * LDA, hi);
#pragma unroll
        for (int cb = 0; cb < 2; ++cb)
            bfg[cb] = frag_k16(Wt + (size_t)(n0 + widN * 32 + cb * 16 + nl) * NC + k0, hi);
#pragma unroll
        for (int rb = 0; rb < 2; ++rb)
#pragma unroll
            for (int cb = 0; cb < 2; ++cb)
                acc[rb][cb] = __builtin_amdgcn_wmma_f32_16x16x32_bf16(
                    false, af[rb], false, bfg[cb], (short)0, acc[rb][cb], false, false);
    }

    // epilogue
#pragma unroll
    for (int rb = 0; rb < 2; ++rb) {
#pragma unroll
        for (int cb = 0; cb < 2; ++cb) {
            int ng = n0 + widN * 32 + cb * 16 + nl;
            if (mode == 0) {
                int which = ng >> 10;
                int rem   = ng & 1023;
                int h     = rem >> 6;
                int d     = rem & 63;
#pragma unroll
                for (int r = 0; r < 8; ++r) {
                    int mg = m0 + widM * 32 + rb * 16 + r + 8 * hi;
                    int bb = mg >> 11;          // /2048
                    int t  = mg & 2047;
                    __bf16 val = f2bf(acc[rb][cb][r]);
                    if (which == 0)
                        qb[(((size_t)(bb * NH + h)) * NT + t) * ND + d] = val;
                    else if (which == 1)
                        kb[(((size_t)(bb * NH + h)) * NT + t) * ND + d] = val;
                    else    // V stored transposed: [B,H,D,T]
                        vt[(((size_t)(bb * NH + h)) * ND + d) * NT + t] = val;
                }
            } else {
                float bv = bias[ng];
#pragma unroll
                for (int r = 0; r < 8; ++r) {
                    int mg = m0 + widM * 32 + rb * 16 + r + 8 * hi;
                    outf[(size_t)mg * NC + ng] = acc[rb][cb][r] + bv;
                }
            }
        }
    }
}

// ---------------------------------------------------------------------------
// Flash attention, one wave per (b, h, 16-query tile).
//   S^T = K * Q^T  and  O^T = V^T * P^T : softmax row (q) == C-fragment lane
//   index -> one shfl_xor(16) per reduction, and the S^T accumulator is
//   bit-for-bit the B-fragment layout of P^T.  V is pre-transposed [B,H,D,T]
//   so every fragment feed is two global_load_b128.
// ---------------------------------------------------------------------------
__global__ __launch_bounds__(256) void attn_kernel(
        const __bf16* __restrict__ Q, const __bf16* __restrict__ K,
        const __bf16* __restrict__ Vt, __bf16* __restrict__ O) {
    const int wid = threadIdx.x >> 5;
    const int lid = threadIdx.x & 31;
    const int gw  = blockIdx.x * (blockDim.x >> 5) + wid;
    if (gw >= NB * NH * (NT / 16)) return;

    const int bh = gw >> 7;            // / (2048/16)
    const int qt = gw & 127;
    const int b  = bh >> 4;
    const int h  = bh & 15;
    const int nl = lid & 15;
    const int hi = lid >> 4;
    const int qrow = qt * 16 + nl;

    const __bf16* Qh  = Q  + (size_t)bh * NT * ND;
    const __bf16* Kh  = K  + (size_t)bh * NT * ND;
    const __bf16* Vth = Vt + (size_t)bh * NT * ND;   // [D][T]

    // B fragments of Q^T (D x 16): element (k=d, n=q) = Q[qrow][d], d-contig
    v16bf bq[2];
#pragma unroll
    for (int c = 0; c < 2; ++c)
        bq[c] = frag_k16(Qh + (size_t)qrow * ND + 32 * c, hi);

    float mstat = -1e30f, lsum = 0.0f;
    v8f o[4] = {};

    const int ngroups = (qt >> 1) + 1;     // key groups of 32, causal
    for (int gi = 0; gi < ngroups; ++gi) {
        const int kg = gi * 32;

        // S^T blocks: A = K rows (16 keys x 32 d), accumulate over D=64
        v8f s[2];
#pragma unroll
        for (int blk = 0; blk < 2; ++blk) {
            int key = kg + blk * 16 + nl;
            v16bf ka0 = frag_k16(Kh + (size_t)key * ND, hi);
            v16bf ka1 = frag_k16(Kh + (size_t)key * ND + 32, hi);
            v8f z = {};
            s[blk] = __builtin_amdgcn_wmma_f32_16x16x32_bf16(false, ka0, false, bq[0],
                                                             (short)0, z, false, false);
            s[blk] = __builtin_amdgcn_wmma_f32_16x16x32_bf16(false, ka1, false, bq[1],
                                                             (short)0, s[blk], false, false);
        }

        // scale + causal mask + row max (row == q == lane&15)
        float rmax = -1e30f;
#pragma unroll
        for (int blk = 0; blk < 2; ++blk)
#pragma unroll
            for (int r = 0; r < 8; ++r) {
                int key = kg + blk * 16 + r + 8 * hi;
                float sv = (key <= qrow) ? s[blk][r] * 0.125f : -1e30f;
                s[blk][r] = sv;
                rmax = fmaxf(rmax, sv);
            }
        rmax = fmaxf(rmax, __shfl_xor(rmax, 16, 32));
        float mnew  = fmaxf(mstat, rmax);
        float alpha = __expf(mstat - mnew);

        float rsum = 0.0f;
        float pv[2][8];
#pragma unroll
        for (int blk = 0; blk < 2; ++blk)
#pragma unroll
            for (int r = 0; r < 8; ++r) {
                float p = __expf(s[blk][r] - mnew);
                pv[blk][r] = p;
                rsum += p;
            }
        rsum += __shfl_xor(rsum, 16, 32);
        lsum  = lsum * alpha + rsum;
        mstat = mnew;

        // P^T as B fragment: dwords 0..3 <- block0 pairs, 4..7 <- block1 pairs
        FragU pbf;
#pragma unroll
        for (int i = 0; i < 4; ++i) {
            pbf.u[i]     = f2bf_pack(pv[0][2 * i], pv[0][2 * i + 1]);
            pbf.u[4 + i] = f2bf_pack(pv[1][2 * i], pv[1][2 * i + 1]);
        }
        v16bf pb = pbf.v;

#pragma unroll
        for (int db = 0; db < 4; ++db)
#pragma unroll
            for (int r = 0; r < 8; ++r) o[db][r] *= alpha;

        // O^T += V^T * P^T : A = V^T (16 d x 32 keys), key-contig in Vt
#pragma unroll
        for (int db = 0; db < 4; ++db) {
            int d = db * 16 + nl;
            v16bf va = frag_k16(Vth + (size_t)d * NT + kg, hi);
            o[db] = __builtin_amdgcn_wmma_f32_16x16x32_bf16(false, va, false, pb,
                                                            (short)0, o[db], false, false);
        }
    }

    const float linv = 1.0f / lsum;
    __bf16* dst = O + ((size_t)(b * NT + qrow)) * NC + h * ND;
#pragma unroll
    for (int db = 0; db < 4; ++db) {
        union { unsigned int u[4]; uint4 v; } u;
#pragma unroll
        for (int i = 0; i < 4; ++i)
            u.u[i] = f2bf_pack(o[db][2 * i] * linv, o[db][2 * i + 1] * linv);
        *(uint4*)(dst + db * 16 + 8 * hi) = u.v;   // 16B-aligned contiguous d
    }
}

// ---------------------------------------------------------------------------
// host-side launcher
// ---------------------------------------------------------------------------
extern "C" void kernel_launch(void* const* d_in, const int* in_sizes, int n_in,
                              void* d_out, int out_size, void* d_ws, size_t ws_size,
                              hipStream_t stream) {
    const float* x      = (const float*)d_in[0];   // [B,T,C]
    const float* w_qkv  = (const float*)d_in[1];   // [C,3C]
    const float* w_proj = (const float*)d_in[2];   // [C,C]
    const float* b_proj = (const float*)d_in[3];   // [C]
    float* out = (float*)d_out;

    char* ws = (char*)d_ws;
    const size_t nX  = (size_t)NM * NC;            // 8.39M
    const size_t nWq = (size_t)NC * 3 * NC;        // 3.15M
    const size_t nWp = (size_t)NC * NC;            // 1.05M
    const size_t nHd = (size_t)NB * NH * NT * ND;  // 8.39M per Q/K/V

    __bf16* Xbf   = (__bf16*)(ws);
    __bf16* WqkvT = (__bf16*)(ws + nX * 2);                         // [3C x C]
    __bf16* WprjT = (__bf16*)(ws + (nX + nWq) * 2);                 // [C x C]
    __bf16* Qb    = (__bf16*)(ws + (nX + nWq + nWp) * 2);           // [B,H,T,D]
    __bf16* Kb    = (__bf16*)(ws + (nX + nWq + nWp + nHd) * 2);     // [B,H,T,D]
    __bf16* Vt    = (__bf16*)(ws + (nX + nWq + nWp + 2 * nHd) * 2); // [B,H,D,T]
    __bf16* Abf   = (__bf16*)(ws + (nX + nWq + nWp + 3 * nHd) * 2); // [B*T, C]

    cvt_bf16_kernel<<<(int)((nX + 255) / 256), 256, 0, stream>>>(x, Xbf, (int)nX);
    cvt_transpose_kernel<<<dim3(NC / 32, 3 * NC / 32), 256, 0, stream>>>(
        w_qkv, WqkvT, NC, 3 * NC);
    cvt_transpose_kernel<<<dim3(NC / 32, NC / 32), 256, 0, stream>>>(
        w_proj, WprjT, NC, NC);

    // QKV projection: [8192,1024] @ [1024,3072] -> Q/K [B,H,T,D], V^T [B,H,D,T]
    gemm_bf16_kernel<<<dim3(3 * NC / BN, NM / BM), 256, 0, stream>>>(
        Xbf, WqkvT, 0, Qb, Kb, Vt, nullptr, nullptr);

    // flash attention: 8192 query tiles of 16, 8 waves per block
    attn_kernel<<<dim3(NB * NH * (NT / 16) / 8), 256, 0, stream>>>(Qb, Kb, Vt, Abf);

    // output projection + bias: [8192,1024] @ [1024,1024] -> fp32 out
    gemm_bf16_kernel<<<dim3(NC / BN, NM / BM), 256, 0, stream>>>(
        Abf, WprjT, 1, nullptr, nullptr, nullptr, out, b_proj);
}
